// EmbeddingLoss_61246233641202
// MI455X (gfx1250) — matrix operand order repaired and checked by reading, not compiled, not run on hardware
//
#include <hip/hip_runtime.h>

// ---------------------------------------------------------------------------
// Contrastive embedding loss, MI455X (gfx1250, wave32, WMMA).
//
// Cost model: gram = E@E^T is 34.4 GFLOP over an 8MB, L2-resident input ->
// pure matrix-core problem. f32 inputs are split into bf16 hi/lo and the
// gram is computed as hi*hi + hi*lo + lo*hi with v_wmma_f32_16x16x32_bf16
// (~f32 accuracy, 3x bf16-WMMA cost). Loss epilogue is fused on the
// accumulators; the 256MB pair matrix never exists in memory. Upper-triangle
// block tiles only (symmetry == the i<j mask).
// ---------------------------------------------------------------------------

#define B_TOTAL 8192
#define D_DIM   256
#define TILE    128                      // workgroup tile (rows == cols)
#define PITCH   264                      // bf16 elems/LDS row: 256 + 8 pad (16B) -> bank-conflict-free
#define NBLK    (B_TOTAL / TILE)         // 64
#define NPAIRS  (NBLK * (NBLK + 1) / 2)  // 2080 upper-tri block tiles

typedef __bf16 bf16_t;
typedef __attribute__((ext_vector_type(16))) __bf16 v16bf;
typedef __attribute__((ext_vector_type(8)))  __bf16 v8bf;
typedef __attribute__((ext_vector_type(8)))  float  v8f;
typedef __attribute__((ext_vector_type(4)))  float  v4f;

// --------------------------- kernel 1: row norms ---------------------------
// One wave per row: lane reads 8 contiguous f32, wave32 shuffle reduce.
// Also zeroes the output accumulator (stream order guarantees it lands
// before kernel 2's atomics).
__global__ __launch_bounds__(256)
void rowsq_kernel(const float* __restrict__ emb, float* __restrict__ sq,
                  float* __restrict__ out) {
    int wave = threadIdx.x >> 5;
    int lane = threadIdx.x & 31;
    int row  = blockIdx.x * 8 + wave;
    const float* p = emb + (size_t)row * D_DIM + lane * 8;
    v4f a = *(const v4f*)p;
    v4f b = *(const v4f*)(p + 4);
    float s = 0.f;
#pragma unroll
    for (int e = 0; e < 4; ++e) s += a[e] * a[e] + b[e] * b[e];
#pragma unroll
    for (int off = 16; off; off >>= 1) s += __shfl_xor(s, off, 32);
    if (lane == 0) sq[row] = s;
    if (blockIdx.x == 0 && threadIdx.x == 0) out[0] = 0.f;
}

// ------------------------- fragment construction ---------------------------
// A-matrix 16x32 bf16 (ISA 7.12.2): lane (0..31), M = lane&15.
//  lanes 0-15 : K = kc+[0..7]  and kc+[16..23]
//  lanes 16-31: K = kc+[8..15] and kc+[24..31]
__device__ __forceinline__ v16bf make_afrag(const bf16_t* p, int half) {
    v8bf x0 = *(const v8bf*)(p + half * 8);        // 16B aligned ds_load_b128
    v8bf x1 = *(const v8bf*)(p + 16 + half * 8);
    v16bf r;
#pragma unroll
    for (int i = 0; i < 8; ++i) { r[i] = x0[i]; r[8 + i] = x1[i]; }
    return r;
}

// B-matrix 32x16 bf16: lane n holds column N=lane&15; lanes 0-15 carry
// K=kc+[0..7],[16..23], lanes 16-31 carry K=kc+[8..15],[24..31] (same as A)
// from row (colBase+N) since B[k,n] = E[colBase+n, k].
__device__ __forceinline__ v16bf make_bfrag(const bf16_t* p, int half) {
    v8bf x0 = *(const v8bf*)(p + half * 8);
    v8bf x1 = *(const v8bf*)(p + 16 + half * 8);
    v16bf r;
#pragma unroll
    for (int i = 0; i < 8; ++i) { r[i] = x0[i]; r[8 + i] = x1[i]; }
    return r;
}

__device__ __forceinline__ v8f wmma_bf16(v16bf a, v16bf b, v8f c) {
    return __builtin_amdgcn_wmma_f32_16x16x32_bf16(
        /*neg_a=*/false, a, /*neg_b=*/false, b,
        /*c_mod=*/(short)0, c, /*reuse_a=*/false, /*reuse_b=*/false);
}

// --------------------------- kernel 2: main tile ---------------------------
__global__ __launch_bounds__(256)
void loss_kernel(const float* __restrict__ emb, const int* __restrict__ labels,
                 const float* __restrict__ sqg, float* __restrict__ out) {
    extern __shared__ __align__(16) char smem[];
    bf16_t* Ihi = (bf16_t*)smem;
    bf16_t* Ilo = Ihi + TILE * PITCH;
    bf16_t* Jhi = Ilo + TILE * PITCH;
    bf16_t* Jlo = Jhi + TILE * PITCH;
    float*  sqI = (float*)(Jlo + TILE * PITCH);
    float*  sqJ = sqI + TILE;
    int*    labI = (int*)(sqJ + TILE);
    int*    labJ = labI + TILE;

    // linear block id -> (bi, bj), bi <= bj  (row bi has NBLK-bi tiles)
    int rem = blockIdx.x, bi = 0;
    while (rem >= NBLK - bi) { rem -= NBLK - bi; ++bi; }
    int bj = bi + rem;
    int rowBase = bi * TILE, colBase = bj * TILE;

    int tid = threadIdx.x;

    // per-tile row norms + labels into LDS
    if (tid < TILE) {
        sqI[tid]  = sqg[rowBase + tid];
        labI[tid] = labels[rowBase + tid];
    } else {
        int u = tid - TILE;
        sqJ[u]  = sqg[colBase + u];
        labJ[u] = labels[colBase + u];
    }

    // Stage I (128 rows) + J (128 rows) panels as split bf16 hi/lo.
    // idx walks [256 rows][64 float4]; consecutive tids -> contiguous 512B
    // global reads per wave (L2-resident stream).
    for (int idx = tid; idx < 256 * (D_DIM / 4); idx += 256) {
        int r  = idx >> 6;
        int c4 = idx & 63;
        int grow = (r < TILE) ? (rowBase + r) : (colBase + (r - TILE));
        v4f v = *(const v4f*)(emb + (size_t)grow * D_DIM + c4 * 4);
        bf16_t* hi = ((r < TILE) ? Ihi : Jhi) + (r & (TILE - 1)) * PITCH + c4 * 4;
        bf16_t* lo = ((r < TILE) ? Ilo : Jlo) + (r & (TILE - 1)) * PITCH + c4 * 4;
#pragma unroll
        for (int e = 0; e < 4; ++e) {
            float  x = v[e];
            bf16_t h = (bf16_t)x;
            hi[e] = h;
            lo[e] = (bf16_t)(x - (float)h);
        }
    }
    __syncthreads();

    // 8 waves: wave tile 64x32 -> 4x2 accumulator tiles of 16x16.
    int wave = tid >> 5, lane = tid & 31;
    int wr = wave & 1, wc = wave >> 1;     // wr:0..1, wc:0..3
    int half = lane >> 4, lr = lane & 15;

    v8f acc[4][2];
#pragma unroll
    for (int m = 0; m < 4; ++m)
#pragma unroll
        for (int n = 0; n < 2; ++n)
#pragma unroll
            for (int e = 0; e < 8; ++e) acc[m][n][e] = 0.f;

    for (int kc = 0; kc < D_DIM; kc += 32) {
        v16bf aH[4], aL[4];
#pragma unroll
        for (int m = 0; m < 4; ++m) {
            int row = wr * 64 + m * 16 + lr;
            aH[m] = make_afrag(Ihi + row * PITCH + kc, half);
            aL[m] = make_afrag(Ilo + row * PITCH + kc, half);
        }
#pragma unroll
        for (int n = 0; n < 2; ++n) {
            int col = wc * 32 + n * 16 + lr;
            v16bf bH = make_bfrag(Jhi + col * PITCH + kc, half);
            v16bf bL = make_bfrag(Jlo + col * PITCH + kc, half);
#pragma unroll
            for (int m = 0; m < 4; ++m) {
                acc[m][n] = wmma_bf16(aH[m], bH, acc[m][n]);  // hi*hi
                acc[m][n] = wmma_bf16(aH[m], bL, acc[m][n]);  // hi*lo
                acc[m][n] = wmma_bf16(aL[m], bH, acc[m][n]);  // lo*hi
            }
        }
    }

    // Fused epilogue. C layout: VGPR e -> M = e + 8*half, N = lane&15.
    const float inv_d = 1.0f / (float)D_DIM;
    const float scale = 1.0f / ((float)B_TOTAL * (float)(B_TOTAL - 1));
    float sum = 0.f;
#pragma unroll
    for (int m = 0; m < 4; ++m)
#pragma unroll
        for (int n = 0; n < 2; ++n)
#pragma unroll
            for (int e = 0; e < 8; ++e) {
                int il = wr * 64 + m * 16 + e + half * 8;
                int jl = wc * 32 + n * 16 + lr;
                int gi = rowBase + il, gj = colBase + jl;
                float gram = acc[m][n][e];
                float mse  = (sqI[il] + sqJ[jl] - 2.0f * gram) * inv_d;
                float v = (labI[il] == labJ[jl]) ? mse : fmaxf(0.f, 1.0f - mse);
                if (gi < gj) sum += v;   // strict upper triangle
            }
    sum *= scale;
#pragma unroll
    for (int off = 16; off; off >>= 1) sum += __shfl_xor(sum, off, 32);
    if (lane == 0) atomicAdd(out, sum);
}

// ------------------------------- launcher ----------------------------------
extern "C" void kernel_launch(void* const* d_in, const int* in_sizes, int n_in,
                              void* d_out, int out_size, void* d_ws, size_t ws_size,
                              hipStream_t stream) {
    (void)in_sizes; (void)n_in; (void)out_size; (void)ws_size;
    const float* emb    = (const float*)d_in[0];
    const int*   labels = (const int*)d_in[1];
    float*       out    = (float*)d_out;
    float*       sq     = (float*)d_ws;   // 8192 floats of scratch

    rowsq_kernel<<<B_TOTAL / 8, 256, 0, stream>>>(emb, sq, out);

    size_t lds = (size_t)4 * TILE * PITCH * sizeof(bf16_t)   // hi/lo panels
               + 2 * TILE * sizeof(float)                    // sqI/sqJ
               + 2 * TILE * sizeof(int);                     // labI/labJ
    loss_kernel<<<NPAIRS, 256, lds, stream>>>(emb, labels, sq, out);
}
